// TwoSelfAttnFuse_11828339933525
// MI455X (gfx1250) — matrix operand order repaired and checked by reading, not compiled, not run on hardware
//
#include <hip/hip_runtime.h>
#include <math.h>

typedef _Float16 v16h __attribute__((ext_vector_type(16)));
typedef _Float16 v8h  __attribute__((ext_vector_type(8)));
typedef float    v8f  __attribute__((ext_vector_type(8)));

#define HEADS 2
#define DH    32
#define NPIX  1600
#define W40   40
#define NWW   20
#define NWIN  400
#define NTOK  16
#define LTOT  6400
#define SCALE 0.17677669529663689f   // 1/sqrt(32)

// workspace byte offsets (all 256-aligned)
#define OFF_QKVX   0u          // 192*1600 f32 = 1228800
#define OFF_QKZ    1228800u    // 128*1600 f32 =  819200
#define OFF_QG16   2048000u    // 2*6400*32 f16 = 819200
#define OFF_QGT16  2867200u    // 2*32*6400 f16 = 819200
#define OFF_OG32   3686400u    // 2*6400*32 f32 = 1638400
#define OFF_CROP   5324800u    // 64*1600 f32  =  409600
// total 5734400 bytes

__device__ __forceinline__ int reflect40(int v) {
  if (v < 0) v = -v;
  if (v >= 40) v = 78 - v;
  return v;
}

// lane-XOR via ds_swizzle (group-of-32 mode: offset = xor<<10 | or<<5 | and)
template <int OFF>
__device__ __forceinline__ float swz(float v) {
  return __int_as_float(__builtin_amdgcn_ds_swizzle(__float_as_int(v), OFF));
}
// max across the 16-lane half-wave (rows are independent per half)
__device__ __forceinline__ float rowmax16(float v) {
  v = fmaxf(v, swz<0x041F>(v));   // xor 1
  v = fmaxf(v, swz<0x081F>(v));   // xor 2
  v = fmaxf(v, swz<0x101F>(v));   // xor 4
  v = fmaxf(v, swz<0x201F>(v));   // xor 8
  return v;
}

__device__ __forceinline__ v16h ones16h() {
  v16h o;
#pragma unroll
  for (int j = 0; j < 16; ++j) o[j] = (_Float16)1.0f;
  return o;
}

// ---------------- Stage 1: fused 1x1 projections ----------------
__global__ void proj_qkv_kernel(const float* __restrict__ x, const float* __restrict__ z,
                                const float* __restrict__ Wqkv, const float* __restrict__ bqkv,
                                const float* __restrict__ Wqkz, const float* __restrict__ bqkz,
                                float* __restrict__ qkvx, float* __restrict__ qkz) {
  int idx = blockIdx.x * blockDim.x + threadIdx.x;   // 320*1600
  if (idx >= 320 * NPIX) return;
  int oc = idx / NPIX;
  int p  = idx % NPIX;
  if (oc < 192) {
    const float* w = Wqkv + oc * 64;
    float acc = bqkv[oc];
#pragma unroll 8
    for (int c = 0; c < 64; ++c) acc += w[c] * x[c * NPIX + p];
    qkvx[oc * NPIX + p] = acc;
  } else {
    int o2 = oc - 192;
    const float* w = Wqkz + o2 * 64;
    float acc = bqkz[o2];
#pragma unroll 8
    for (int c = 0; c < 64; ++c) acc += w[c] * z[c * NPIX + p];
    qkz[o2 * NPIX + p] = acc;
  }
}

// ---------------- Stage 2: window attention (one wave per window*head) ----------------
__global__ __launch_bounds__(32) void win_attn_kernel(
    const float* __restrict__ qkvx, const float* __restrict__ qkz,
    const float* __restrict__ pbeta,
    _Float16* __restrict__ qg16, _Float16* __restrict__ qgT16) {
  __shared__ float lds[16 * 17];
  const int lane = threadIdx.x;
  const int row  = lane & 15;
  const int hi   = lane >> 4;
  const int win  = blockIdx.x >> 1;
  const int head = blockIdx.x & 1;
  const int wy = win / NWW, wx = win % NWW;

  float pb = pbeta[0];
  float beta = (pb > 20.f ? pb : log1pf(__expf(pb))) + 1e-6f;
  float sb = SCALE * beta;

  // pixel index for each of the 16 window tokens (reflect padding, uniform per block)
  int ptab[16];
#pragma unroll
  for (int t = 0; t < 16; ++t) {
    int pr = reflect40(wy * 2 + (t >> 2) - 1);
    int pc = reflect40(wx * 2 + (t & 3) - 1);
    ptab[t] = pr * W40 + pc;
  }
  const int myp = ptab[row];

  // A operands (qx,qz): lane holds Q row=token, elem j -> K dim 8*hi + 16*(j>>3) + (j&7)
  // B operands (kx,kz): lane holds K^T col=token, elem j -> K dim 8*hi + 16*(j>>3) + (j&7)
  // qx pre-scaled by 1/sqrt(d); z-term scale folded into sb.
  v16h aqx, aqz, bkx, bkz;
#pragma unroll
  for (int j = 0; j < 16; ++j) {
    int kA = 8 * hi + ((j & 8) << 1) + (j & 7);
    int cA = head * DH + kA;
    aqx[j] = (_Float16)(qkvx[cA * NPIX + myp] * SCALE);
    aqz[j] = (_Float16)qkz[cA * NPIX + myp];
    int kB = 8 * hi + ((j & 8) << 1) + (j & 7);
    int cB = head * DH + kB;
    bkx[j] = (_Float16)qkvx[(64 + cB) * NPIX + myp];
    bkz[j] = (_Float16)qkz[(64 + cB) * NPIX + myp];
  }

  v8f zero8 = {0.f, 0.f, 0.f, 0.f, 0.f, 0.f, 0.f, 0.f};
  v8f sx = __builtin_amdgcn_wmma_f32_16x16x32_f16(false, aqx, false, bkx, (short)0, zero8, false, false);
  v8f sz = __builtin_amdgcn_wmma_f32_16x16x32_f16(false, aqz, false, bkz, (short)0, zero8, false, false);

  // combined logits; softmax is shift-invariant -> subtract combined row max
#pragma unroll
  for (int r = 0; r < 8; ++r) {
    float t = sx[r] + sb * sz[r];
    float m = rowmax16(t);
    lds[(r + 8 * hi) * 17 + row] = __expf(t - m);   // unnormalized probs
  }
  __syncthreads();

  // P (16x16) as A operand, K=16..31 zero-padded
  v16h ap;
#pragma unroll
  for (int j = 0; j < 16; ++j) {
    if (!hi) ap[j] = (_Float16)lds[row * 17 + j];
    else     ap[j] = (_Float16)0.f;
  }

  // row sums via WMMA against all-ones B (replicated across lanes, row-aligned per vgpr)
  v8f rs = __builtin_amdgcn_wmma_f32_16x16x32_f16(false, ap, false, ones16h(), (short)0, zero8, false, false);

  // O = P @ V, two 16-wide dim tiles; V rows 16..31 are zero (P padded anyway)
  v8f o[2];
#pragma unroll
  for (int nt = 0; nt < 2; ++nt) {
    int dim = row + 16 * nt;
    int cV = head * DH + dim;
    v16h bv;
#pragma unroll
    for (int j = 0; j < 16; ++j) {
      float val = qkvx[(128 + cV) * NPIX + ptab[j]];
      bv[j] = hi ? (_Float16)0.f : (_Float16)val;
    }
    o[nt] = __builtin_amdgcn_wmma_f32_16x16x32_f16(false, ap, false, bv, (short)0, zero8, false, false);
  }

  // epilogue: normalize and write qg in f16 (row-major) and transposed f16
  float inv8[8];
#pragma unroll
  for (int r = 0; r < 8; ++r) inv8[r] = 1.0f / rs[r];
#pragma unroll
  for (int nt = 0; nt < 2; ++nt) {
#pragma unroll
    for (int r = 0; r < 8; ++r) {
      int M = r + 8 * hi;
      int dim = row + 16 * nt;
      int Lg = win * NTOK + M;
      _Float16 v = (_Float16)(o[nt][r] * inv8[r]);
      qg16[(size_t)(head * LTOT + Lg) * DH + dim] = v;
      qgT16[(size_t)(head * DH + dim) * LTOT + Lg] = v;
    }
  }
}

// ---------------- Stage 3: global flash attention (one wave per 16-query tile) ----------------
__global__ __launch_bounds__(32) void global_attn_kernel(
    const _Float16* __restrict__ qg16, const _Float16* __restrict__ qgT16,
    float* __restrict__ og32) {
  __shared__ float lds[16 * 33];
  const int lane  = threadIdx.x;
  const int row   = lane & 15;
  const int hi    = lane >> 4;
  const int qtile = blockIdx.x % 400;
  const int head  = blockIdx.x / 400;

  // Q tile, A layout, pre-scaled by 1/sqrt(d)
  const _Float16* qrow = qg16 + (size_t)(head * LTOT + qtile * 16 + row) * DH;
  v16h aq;
  {
    v8h lo  = *reinterpret_cast<const v8h*>(qrow + 16 * hi);
    v8h hi8 = *reinterpret_cast<const v8h*>(qrow + 16 * hi + 8);
#pragma unroll
    for (int j = 0; j < 8; ++j) {
      aq[j]     = (_Float16)((float)lo[j]  * SCALE);
      aq[8 + j] = (_Float16)((float)hi8[j] * SCALE);
    }
  }
  const v16h bones = ones16h();

  v8f o0 = {0.f,0.f,0.f,0.f,0.f,0.f,0.f,0.f};
  v8f o1 = o0;
  float m8[8], l8[8];
#pragma unroll
  for (int r = 0; r < 8; ++r) { m8[r] = -1e30f; l8[r] = 0.f; }

  for (int kt = 0; kt < 200; ++kt) {
    const _Float16* k0 = qg16 + (size_t)(head * LTOT + kt * 32 + row) * DH + 16 * hi;
    const _Float16* k1 = qg16 + (size_t)(head * LTOT + kt * 32 + 16 + row) * DH + 16 * hi;
    if (kt < 196) {   // uniform condition: prefetch upcoming K/V lines
      __builtin_prefetch(k0 + 4 * 32 * DH, 0, 1);
      __builtin_prefetch(qgT16 + (size_t)(head * DH + row) * LTOT + (kt + 4) * 32, 0, 1);
    }
    // two key tiles (32 keys), B layout: contiguous dims per lane
    v16h bk0, bk1;
    {
      v8h a0 = *reinterpret_cast<const v8h*>(k0);
      v8h a1 = *reinterpret_cast<const v8h*>(k0 + 8);
      v8h b0 = *reinterpret_cast<const v8h*>(k1);
      v8h b1 = *reinterpret_cast<const v8h*>(k1 + 8);
#pragma unroll
      for (int j = 0; j < 8; ++j) {
        bk0[j] = a0[j]; bk0[8 + j] = a1[j];
        bk1[j] = b0[j]; bk1[8 + j] = b1[j];
      }
    }
    v8f zero8 = {0.f,0.f,0.f,0.f,0.f,0.f,0.f,0.f};
    v8f s0 = __builtin_amdgcn_wmma_f32_16x16x32_f16(false, aq, false, bk0, (short)0, zero8, false, false);
    v8f s1 = __builtin_amdgcn_wmma_f32_16x16x32_f16(false, aq, false, bk1, (short)0, zero8, false, false);

    // online softmax: row max via ds_swizzle reduction, rescale running state
#pragma unroll
    for (int r = 0; r < 8; ++r) {
      float cur = rowmax16(fmaxf(s0[r], s1[r]));
      float mn = fmaxf(m8[r], cur);
      float fr = __expf(m8[r] - mn);
      m8[r] = mn;
      float p0 = __expf(s0[r] - mn);
      float p1 = __expf(s1[r] - mn);
      l8[r] *= fr;
      o0[r] *= fr; o1[r] *= fr;
      lds[(r + 8 * hi) * 33 + row]      = p0;
      lds[(r + 8 * hi) * 33 + 16 + row] = p1;
    }
    __syncthreads();
    v16h ap;
#pragma unroll
    for (int j = 0; j < 16; ++j) {
      int k = 16 * hi + j;
      ap[j] = (_Float16)lds[row * 33 + k];
    }
    __syncthreads();

    // row sums via WMMA against ones; l += rowsum
    v8f rs = __builtin_amdgcn_wmma_f32_16x16x32_f16(false, ap, false, bones, (short)0, zero8, false, false);
#pragma unroll
    for (int r = 0; r < 8; ++r) l8[r] += rs[r];

    // O += P @ V ; V streamed contiguously from transposed f16 buffer
#pragma unroll
    for (int nt = 0; nt < 2; ++nt) {
      int dim = row + 16 * nt;
      const _Float16* vp = qgT16 + (size_t)(head * DH + dim) * LTOT + kt * 32 + 16 * hi;
      v8h a0 = *reinterpret_cast<const v8h*>(vp);
      v8h a1 = *reinterpret_cast<const v8h*>(vp + 8);
      v16h bv;
#pragma unroll
      for (int j = 0; j < 8; ++j) { bv[j] = a0[j]; bv[8 + j] = a1[j]; }
      if (nt == 0) o0 = __builtin_amdgcn_wmma_f32_16x16x32_f16(false, ap, false, bv, (short)0, o0, false, false);
      else         o1 = __builtin_amdgcn_wmma_f32_16x16x32_f16(false, ap, false, bv, (short)0, o1, false, false);
    }
  }

#pragma unroll
  for (int r = 0; r < 8; ++r) {
    float inv = 1.0f / l8[r];
    int Lg = qtile * 16 + r + 8 * hi;
    og32[(size_t)(head * LTOT + Lg) * DH + row]      = o0[r] * inv;
    og32[(size_t)(head * LTOT + Lg) * DH + 16 + row] = o1[r] * inv;
  }
}

// ---------------- Stage 4a: overlap-add + count average + crop ----------------
__global__ void overlap_kernel(const float* __restrict__ og32, float* __restrict__ crop) {
  int idx = blockIdx.x * blockDim.x + threadIdx.x;   // 64*1600
  if (idx >= 64 * NPIX) return;
  int c = idx / NPIX;
  int p = idx % NPIX;
  int y = p / W40, x = p % W40;
  int yp = y + 1, xp = x + 1;                        // padded coords
  int wy_lo = (yp - 2) / 2; if (wy_lo < 0) wy_lo = 0;
  int wy_hi = yp / 2;       if (wy_hi > 19) wy_hi = 19;
  int wx_lo = (xp - 2) / 2; if (wx_lo < 0) wx_lo = 0;
  int wx_hi = xp / 2;       if (wx_hi > 19) wx_hi = 19;
  int h = c >> 5, dd = c & 31;
  float sum = 0.f; int cnt = 0;
  for (int wy = wy_lo; wy <= wy_hi; ++wy)
    for (int wx = wx_lo; wx <= wx_hi; ++wx) {
      int t = (yp - 2 * wy) * 4 + (xp - 2 * wx);
      int Lg = (wy * NWW + wx) * NTOK + t;
      sum += og32[(size_t)(h * LTOT + Lg) * DH + dd];
      cnt++;
    }
  crop[c * NPIX + p] = sum / ((float)cnt + 1e-6f);
}

// ---------------- Stage 4b: proj conv + gated residual ----------------
__global__ void final_kernel(const float* __restrict__ crop, const float* __restrict__ xfeat,
                             const float* __restrict__ Wproj, const float* __restrict__ bproj,
                             const float* __restrict__ pa, float* __restrict__ out) {
  int idx = blockIdx.x * blockDim.x + threadIdx.x;   // 64*1600
  if (idx >= 64 * NPIX) return;
  int oc = idx / NPIX;
  int p  = idx % NPIX;
  float a = 2.f / (1.f + __expf(-pa[0]));
  const float* w = Wproj + oc * 64;
  float acc = bproj[oc];
#pragma unroll 8
  for (int c = 0; c < 64; ++c) acc += w[c] * crop[c * NPIX + p];
  out[oc * NPIX + p] = xfeat[oc * NPIX + p] + a * acc;
}

extern "C" void kernel_launch(void* const* d_in, const int* in_sizes, int n_in,
                              void* d_out, int out_size, void* d_ws, size_t ws_size,
                              hipStream_t stream) {
  const float* x     = (const float*)d_in[0];
  const float* z     = (const float*)d_in[1];
  const float* Wqkv  = (const float*)d_in[2];
  const float* bqkv  = (const float*)d_in[3];
  const float* Wqkz  = (const float*)d_in[4];
  const float* bqkz  = (const float*)d_in[5];
  const float* Wproj = (const float*)d_in[6];
  const float* bproj = (const float*)d_in[7];
  const float* pa    = (const float*)d_in[8];
  const float* pbeta = (const float*)d_in[9];

  char* ws = (char*)d_ws;
  float*    qkvx  = (float*)(ws + OFF_QKVX);
  float*    qkz   = (float*)(ws + OFF_QKZ);
  _Float16* qg16  = (_Float16*)(ws + OFF_QG16);
  _Float16* qgT16 = (_Float16*)(ws + OFF_QGT16);
  float*    og32  = (float*)(ws + OFF_OG32);
  float*    crop  = (float*)(ws + OFF_CROP);
  float*    out   = (float*)d_out;

  proj_qkv_kernel<<<(320 * NPIX + 255) / 256, 256, 0, stream>>>(x, z, Wqkv, bqkv, Wqkz, bqkz, qkvx, qkz);
  win_attn_kernel<<<NWIN * HEADS, 32, 0, stream>>>(qkvx, qkz, pbeta, qg16, qgT16);
  global_attn_kernel<<<HEADS * 400, 32, 0, stream>>>(qg16, qgT16, og32);
  overlap_kernel<<<(64 * NPIX + 255) / 256, 256, 0, stream>>>(og32, crop);
  final_kernel<<<(64 * NPIX + 255) / 256, 256, 0, stream>>>(crop, x, Wproj, bproj, pa, out);
}
